// VQEMA_18408229830940
// MI455X (gfx1250) — matrix operand order repaired and checked by reading, not compiled, not run on hardware
//
#include <hip/hip_runtime.h>
#include <hip/hip_bf16.h>

// Shapes from the reference
#define B_ 8
#define C_ 256    // N_IN
#define D_ 128
#define K_ 1024
#define N_ 4096
#define GAMMA 0.99f
#define ONE_M_GAMMA 0.01f

typedef __attribute__((ext_vector_type(16))) __bf16 v16bf;
typedef __attribute__((ext_vector_type(8)))  __bf16 v8bf;
typedef __attribute__((ext_vector_type(8)))  float  v8f;
typedef __attribute__((ext_vector_type(4)))  int    v4i;

#if __has_builtin(__builtin_amdgcn_global_load_async_to_lds_b128)
#define HAVE_ASYNC 1
typedef __attribute__((address_space(1))) v4i* as1_v4i_p;
typedef __attribute__((address_space(3))) v4i* as3_v4i_p;
#endif

// lane <-> lane^16 exchange (SWAPX16): xor=0x10, or=0, and=0x1f -> 0x401f
static __device__ __forceinline__ float swz16f(float x) {
  int i = __builtin_amdgcn_ds_swizzle(__builtin_bit_cast(int, x), 0x401f);
  return __builtin_bit_cast(float, i);
}
static __device__ __forceinline__ int swz16i(int x) {
  return __builtin_amdgcn_ds_swizzle(x, 0x401f);
}

// 16-byte global->LDS tile mover: async on gfx1250 toolchains that expose the
// builtin (tracked by ASYNCcnt), synchronous b128 copy otherwise.
static __device__ __forceinline__ void tile_ld16(__bf16* lds_dst, const __bf16* gsrc) {
#if HAVE_ASYNC
  __builtin_amdgcn_global_load_async_to_lds_b128((as1_v4i_p)gsrc, (as3_v4i_p)lds_dst, 0, 0);
#else
  *(v8bf*)lds_dst = *(const v8bf*)gsrc;
#endif
}
static __device__ __forceinline__ void tile_wait() {
#if HAVE_ASYNC
#if __has_builtin(__builtin_amdgcn_s_wait_asynccnt)
  __builtin_amdgcn_s_wait_asynccnt(0);
#else
  asm volatile("s_wait_asynccnt 0" ::: "memory");
#endif
#endif
}

// ---------------------------------------------------------------------------
// Kernel 1: ze[b,d,n] = sum_c W[d,c] * z[b,c,n]   (bf16 WMMA, f32 accum)
// Also emits zebf(B,N,D) bf16: per-column contiguous copy for the argmin GEMM.
// One wave = 16(d) x 16(n) tile; 8 waves/block cover all D=128.
// grid = B * N/16 blocks of 256 threads.
// ---------------------------------------------------------------------------
__global__ __launch_bounds__(256) void k_ze(const float* __restrict__ z,
                                            const float* __restrict__ W,
                                            float* __restrict__ ze,
                                            __bf16* __restrict__ zebf) {
  const int lane = threadIdx.x & 31;
  const int wave = threadIdx.x >> 5;
  const int m = lane & 15;     // A row / B col / D col index for this lane
  const int h = lane >> 4;     // half select
  const int b  = blockIdx.x >> 8;            // 256 n-tiles per batch
  const int n0 = (blockIdx.x & 255) << 4;
  const int d0 = wave << 4;
  const int n  = n0 + m;

  v8f acc = {};
  for (int c0 = 0; c0 < C_; c0 += 32) {
    // A fragment: W_lin[d0+m][c0 + kk], kk = e+8h (e<8) / 16+(e-8)+8h (e>=8)
    v16bf a;
    const float* wrow = W + (size_t)(d0 + m) * C_ + c0;
#pragma unroll
    for (int e = 0; e < 8; ++e) a[e] = (__bf16)wrow[e + 8 * h];
#pragma unroll
    for (int e = 0; e < 8; ++e) a[8 + e] = (__bf16)wrow[16 + 8 * h + e];
    // B fragment: z[b][c0+kk][n], kk = t+8h (t<8) / 16+(t-8)+8h (t>=8)
    v16bf bb;
    const float* zcol = z + ((size_t)b * C_ + c0 + 8 * h) * N_ + n;
#pragma unroll
    for (int t = 0; t < 16; ++t) bb[t] = (__bf16)zcol[(size_t)(t < 8 ? t : t + 8) * N_];
    acc = __builtin_amdgcn_wmma_f32_16x16x32_bf16(false, a, false, bb,
                                                  (short)0, acc, false, false);
  }
  // C/D layout: VGPR r -> row d = d0 + r + 8h, lane column n
  v8bf pk;
#pragma unroll
  for (int r = 0; r < 8; ++r) {
    int dd = d0 + r + 8 * h;
    ze[((size_t)b * D_ + dd) * N_ + n] = acc[r];
    pk[r] = (__bf16)acc[r];
  }
  // transposed bf16 copy: 8 consecutive d at one shot (16B store)
  *(v8bf*)&zebf[((size_t)b * N_ + n) * D_ + d0 + 8 * h] = pk;
}

// ---------------------------------------------------------------------------
// Kernel 2a: embbf[i] = bf16(emb[i])   (one-time codebook conversion)
// ---------------------------------------------------------------------------
__global__ __launch_bounds__(256) void k_cvt(const float* __restrict__ emb,
                                             __bf16* __restrict__ embbf) {
  int i = blockIdx.x * 256 + threadIdx.x;
  if (i < K_ * D_) embbf[i] = (__bf16)emb[i];
}

// ---------------------------------------------------------------------------
// Kernel 2b: qpack[k] = (||emb[k]||^2, ||emb[k]||)
// ---------------------------------------------------------------------------
__global__ __launch_bounds__(256) void k_qstats(const float* __restrict__ emb,
                                                float2* __restrict__ qpack) {
  int k = blockIdx.x * 256 + threadIdx.x;
  if (k < K_) {
    const float* row = emb + (size_t)k * D_;
    float s = 0.f;
#pragma unroll 8
    for (int d = 0; d < D_; ++d) s += row[d] * row[d];
    qpack[k] = make_float2(s, sqrtf(s));
  }
}

// ---------------------------------------------------------------------------
// Kernel 3: fused dots-GEMM + scaled-L2 + argmin over K.
// One wave owns 16 ze columns (B-fragments loop-invariant in registers);
// 8 waves/block share a double-buffered 16x128 bf16 emb tile in LDS, fed by
// async global->LDS b128 transfers overlapped with the WMMA epilogue.
// grid = B * N/128 blocks of 256 threads.
// ---------------------------------------------------------------------------
__global__ __launch_bounds__(256) void k_argmin(const __bf16* __restrict__ zebf,
                                                const __bf16* __restrict__ embbf,
                                                const float2* __restrict__ qpack,
                                                int* __restrict__ min_ind) {
  __shared__ __align__(16) __bf16 sA[2][16 * D_];

  const int lane = threadIdx.x & 31;
  const int wave = threadIdx.x >> 5;
  const int m = lane & 15;
  const int h = lane >> 4;
  const int b  = blockIdx.x >> 5;                      // 32 blocks per batch
  const int n0 = ((blockIdx.x & 31) << 7) + (wave << 4);
  const int n  = n0 + m;

  // B fragments straight from the transposed bf16 ze: t<8 -> d = 32dc+8h+t ; t>=8 -> d = 32dc+16+8h+(t-8)
  v16bf bfrag[4];
  float zs = 0.f;
  const __bf16* zrow = zebf + ((size_t)b * N_ + n) * D_;
#pragma unroll
  for (int dc = 0; dc < 4; ++dc) {
    v8bf b0 = *(const v8bf*)&zrow[32 * dc + 8 * h];
    v8bf b1 = *(const v8bf*)&zrow[32 * dc + 16 + 8 * h];
#pragma unroll
    for (int i = 0; i < 8; ++i) {
      float f0 = (float)b0[i], f1 = (float)b1[i];
      zs += f0 * f0 + f1 * f1;
      bfrag[dc][i] = b0[i];
      bfrag[dc][8 + i] = b1[i];
    }
  }
  zs += swz16f(zs);           // combine the two halves of the d-range
  const float zn = sqrtf(zs);

  float best  = 3.4e38f;
  int   bestk = 0;

  // prologue: start fetching tile 0 (16 rows x 128 cols bf16 = 4KB; 16B/thread)
  tile_ld16(&sA[0][threadIdx.x * 8], embbf + threadIdx.x * 8);

  for (int k0 = 0; k0 < K_; k0 += 16) {
    const int buf = (k0 >> 4) & 1;
    tile_wait();              // my async transfer into sA[buf] landed
    __syncthreads();          // everyone's landed; prior readers of buf^1 done
    if (k0 + 16 < K_)         // overlap next tile with this tile's math
      tile_ld16(&sA[buf ^ 1][threadIdx.x * 8],
                embbf + (size_t)(k0 + 16) * D_ + threadIdx.x * 8);

    v8f acc = {};
#pragma unroll
    for (int dc = 0; dc < 4; ++dc) {
      // A layout: elems 0..7 -> d = 32dc + 8h + e ; 8..15 -> d = 32dc+16+8h+(e-8)
      const v8bf a0 = *(const v8bf*)&sA[buf][m * D_ + 32 * dc + 8 * h];
      const v8bf a1 = *(const v8bf*)&sA[buf][m * D_ + 32 * dc + 16 + 8 * h];
      v16bf a;
#pragma unroll
      for (int i = 0; i < 8; ++i) { a[i] = a0[i]; a[8 + i] = a1[i]; }
      acc = __builtin_amdgcn_wmma_f32_16x16x32_bf16(false, a, false, bfrag[dc],
                                                    (short)0, acc, false, false);
    }

    // argmin epilogue: this lane holds rows k = k0 + 8h + r for column n.
    const float2* qp = qpack + k0 + 8 * h;
    float bv = 3.4e38f;
    int   bk = k0 + 8 * h;
#pragma unroll
    for (int r = 0; r < 8; ++r) {
      float2 q   = qp[r];
      float dist = zs + q.x - 2.0f * acc[r];
      dist = dist > 0.f ? dist : 0.f;
      float den = zn + q.y;
      float val = dist / (den * den);       // monotone in snorm -> same argmin
      if (val < bv) { bv = val; bk = k0 + 8 * h + r; }
    }
    // merge with lane^16 (other 8 rows of same column); keep smallest index on tie
    float ov = swz16f(bv);
    int   ok = swz16i(bk);
    if (ov < bv || (ov == bv && ok < bk)) { bv = ov; bk = ok; }
    if (bv < best || (bv == best && bk < bestk)) { best = bv; bestk = bk; }
  }

  if (h == 0) min_ind[(size_t)b * N_ + n] = bestk;
}

// ---------------------------------------------------------------------------
// Kernel 4: seed EMA outputs with gamma * old value (scatter adds on top)
// ---------------------------------------------------------------------------
__global__ __launch_bounds__(256) void k_init_ema(const float* __restrict__ numer_in,
                                                  const float* __restrict__ denom_in,
                                                  float* __restrict__ numer_out,
                                                  float* __restrict__ denom_out) {
  int i = blockIdx.x * 256 + threadIdx.x;
  if (i < K_ * D_) numer_out[i] = GAMMA * numer_in[i];
  if (i < K_)      denom_out[i] = GAMMA * denom_in[i];
}

// ---------------------------------------------------------------------------
// Kernel 5: gather zq output + EMA scatter-add (fused).
// One thread per (b,n) column; loops over d. grid = B*N/256.
// ---------------------------------------------------------------------------
__global__ __launch_bounds__(256) void k_out(const float* __restrict__ emb,
                                             const float* __restrict__ ze,
                                             const int* __restrict__ min_ind,
                                             float* __restrict__ zq,
                                             float* __restrict__ numer_out,
                                             float* __restrict__ denom_out) {
  int cidx = blockIdx.x * 256 + threadIdx.x;   // [0, B*N)
  int b = cidx >> 12;                          // /4096
  int n = cidx & (N_ - 1);
  int idx = min_ind[cidx];
  const float* erow = emb + (size_t)idx * D_;
#pragma unroll 4
  for (int d = 0; d < D_; ++d) {
    float ev = erow[d];
    size_t off = ((size_t)b * D_ + d) * N_ + n;
    zq[off] = ev;                              // straight-through forward == zq
    atomicAdd(&numer_out[(size_t)idx * D_ + d], ONE_M_GAMMA * ze[off]);
  }
  atomicAdd(&denom_out[idx], ONE_M_GAMMA);
}

// ---------------------------------------------------------------------------
extern "C" void kernel_launch(void* const* d_in, const int* in_sizes, int n_in,
                              void* d_out, int out_size, void* d_ws, size_t ws_size,
                              hipStream_t stream) {
  const float* z         = (const float*)d_in[0];  // (B, C, N)
  const float* W         = (const float*)d_in[1];  // (D, C)
  const float* emb       = (const float*)d_in[2];  // (K, D)
  const float* ema_numer = (const float*)d_in[3];  // (K, D)
  const float* ema_denom = (const float*)d_in[4];  // (K,)

  float* out       = (float*)d_out;
  float* zq        = out;                                   // (B, D, N)
  float* numer_out = out + (size_t)B_ * D_ * N_;            // (K, D)
  float* denom_out = numer_out + (size_t)K_ * D_;           // (K,)

  float*  ze      = (float*)d_ws;                           // (B, D, N) fp32
  __bf16* zebf    = (__bf16*)(ze + (size_t)B_ * D_ * N_);   // (B, N, D) bf16
  __bf16* embbf   = zebf + (size_t)B_ * N_ * D_;            // (K, D) bf16
  float2* qpack   = (float2*)(embbf + (size_t)K_ * D_);     // (K,) {qsq, qnorm}
  int*    min_ind = (int*)(qpack + K_);                     // (B*N,)

  k_cvt     <<<(K_ * D_ + 255) / 256, 256, 0, stream>>>(emb, embbf);
  k_qstats  <<<(K_ + 255) / 256,      256, 0, stream>>>(emb, qpack);
  k_ze      <<<B_ * (N_ / 16),        256, 0, stream>>>(z, W, ze, zebf);
  k_argmin  <<<B_ * (N_ / 128),       256, 0, stream>>>(zebf, embbf, qpack, min_ind);
  k_init_ema<<<(K_ * D_ + 255) / 256, 256, 0, stream>>>(ema_numer, ema_denom,
                                                        numer_out, denom_out);
  k_out     <<<(B_ * N_) / 256,       256, 0, stream>>>(emb, ze, min_ind, zq,
                                                        numer_out, denom_out);
}
